// Attention_50757923504468
// MI455X (gfx1250) — matrix operand order repaired and checked by reading, not compiled, not run on hardware
//
#include <hip/hip_runtime.h>
#include <hip/hip_bf16.h>
#include <cmath>
#include <cstdint>

// Problem dims (fixed by the reference): B=32, S=4096, QD=VD=HD=512
#define NBATCH 32
#define SEQ    4096
#define DVD    512
#define DHD    512

typedef __attribute__((ext_vector_type(16))) __bf16       v16bf;
typedef __attribute__((ext_vector_type(8)))  float        v8f;
typedef __attribute__((ext_vector_type(4)))  __bf16       bf16x4;
typedef __attribute__((ext_vector_type(4)))  unsigned int u32x4;
typedef __attribute__((ext_vector_type(8)))  int          i32x8;
typedef __attribute__((ext_vector_type(4)))  int          i32x4;

#if __has_builtin(__builtin_amdgcn_tensor_load_to_lds)
#define USE_TDM 1
#endif

// ---------------------------------------------------------------------------
// Kernel 0a: qb[b][h] = bq[h] + sum_v query[b][v] * Wq[v][h]   (tiny GEMM)
// ---------------------------------------------------------------------------
__global__ void qproj_kernel(const float* __restrict__ query,
                             const float* __restrict__ Wq,
                             const float* __restrict__ bq,
                             float* __restrict__ qb) {
  const int b = blockIdx.x;
  const float* qrow = query + (size_t)b * DVD;
  for (int h = threadIdx.x; h < DHD; h += blockDim.x) {
    float acc = bq[h];
#pragma unroll 8
    for (int v = 0; v < DVD; ++v)
      acc = fmaf(qrow[v], Wq[(size_t)v * DHD + h], acc);
    qb[(size_t)b * DHD + h] = acc;
  }
}

// ---------------------------------------------------------------------------
// Kernel 0b: one-time Wk f32 -> bf16 (512KB, L2-resident for all score WGs)
// ---------------------------------------------------------------------------
__global__ void wk_bf16_kernel(const float* __restrict__ Wk,
                               __bf16* __restrict__ wkbf) {
  const int i = (blockIdx.x * 256 + threadIdx.x) * 4;  // 256 blocks cover 512*512
  float4 f = *(const float4*)(Wk + i);
  bf16x4 o;
  o[0] = (__bf16)f.x; o[1] = (__bf16)f.y; o[2] = (__bf16)f.z; o[3] = (__bf16)f.w;
  *(bf16x4*)(wkbf + i) = o;
}

// ---------------------------------------------------------------------------
// TDM: issue one 2D tile load  Wk_bf16[:, n0:n0+16] -> LDS  (padded rows)
// D# layout per CDNA5 ISA.
//   tensor: 512 x 512 bf16 row-major (dim0 = contiguous h, stride 512)
//   tile:   tile_dim0=16 (h), tile_dim1=512 (k)
//   LDS pad: after every 8 DWORDs (32B row) insert 4 DWORDs (16B)
//            -> row stride 48B == Bs[k][24] bf16 layout used by the WMMA reads
// ---------------------------------------------------------------------------
__device__ __forceinline__ void tdm_load_wk_tile(const __bf16* wkbf, int n0,
                                                 unsigned int lds_byte_addr) {
#ifdef USE_TDM
  const unsigned long long gaddr = (unsigned long long)(uintptr_t)(wkbf + n0);
  u32x4 g0;
  g0[0] = 1u;                                   // count=1 (valid user D#)
  g0[1] = lds_byte_addr;                        // lds_addr
  g0[2] = (unsigned int)gaddr;                  // global_addr[31:0]
  g0[3] = (unsigned int)((gaddr >> 32) & 0x01FFFFFFu) | (2u << 30);  // addr[56:32] | type=2

  i32x8 g1;
  g1[0] = (int)((1u << 16)      // data_size = 2 bytes
              | (1u << 20)      // pad_enable
              | (2u << 22)      // pad_interval: 8 DWORDs
              | (3u << 25));    // pad_amount:   4 DWORDs
  g1[1] = (int)(512u << 16);    // tensor_dim0[15:0] = 512 (bits 79:48)
  g1[2] = (int)(512u << 16);    // tensor_dim1[15:0] = 512 (bits 111:80)
  g1[3] = (int)(16u  << 16);    // tile_dim0 = 16 (bits 127:112)
  g1[4] = 512;                  // tile_dim1 = 512 (bits 143:128); tile_dim2 = 0
  g1[5] = 512;                  // tensor_dim0_stride[31:0] = 512 (bits 207:160)
  g1[6] = 0;
  g1[7] = 0;                    // tensor_dim1_stride = 0 (unused for 2D)

  i32x4 z4 = {0, 0, 0, 0};      // groups 2/3: 2D tensor -> unused
#if __clang_major__ >= 23
  i32x8 z8 = {0, 0, 0, 0, 0, 0, 0, 0};
  __builtin_amdgcn_tensor_load_to_lds(g0, g1, z4, z4, z8, 0);
#else
  __builtin_amdgcn_tensor_load_to_lds(g0, g1, z4, z4, 0);
#endif
#else
  (void)wkbf; (void)n0; (void)lds_byte_addr;
#endif
}

// ---------------------------------------------------------------------------
// Kernel 1: fused  k = value@Wk ; score = Wo . tanh(qb + k) + bo  (masked)
// 8 waves/WG, 128 rows. A (value rows, bf16) lives in registers; Wk tiles are
// DMA'd into double-buffered LDS by the Tensor Data Mover, overlapped with
// the WMMA stream of the previous tile.
// ---------------------------------------------------------------------------
__global__ __launch_bounds__(256) void score_kernel(
    const float* __restrict__ value,
    const unsigned char* __restrict__ mask,
    const __bf16* __restrict__ wkbf,
    const float* __restrict__ Wo,
    const float* __restrict__ bo_p,
    const float* __restrict__ qb,
    float* __restrict__ scores) {

  __shared__ __bf16 Bs[2][512][24];   // double-buffered Wk tile, 48B row stride
  __shared__ float  qs[DHD];
  __shared__ float  wos[DHD];

  const int tid  = threadIdx.x;
  const int lane = tid & 31;
  const int wave = tid >> 5;

  const int blk_row0 = blockIdx.x * 128;       // 1024 blocks * 128 rows = B*S
  const int b        = blk_row0 / SEQ;         // one batch per WG
  const int m0       = blk_row0 + wave * 16;   // this wave's 16 rows

  for (int i = tid; i < DHD; i += 256) {
    qs[i]  = qb[(size_t)b * DHD + i];
    wos[i] = Wo[i];
  }

  // --- load 16 rows of value -> bf16 A fragments in registers -------------
  // 16-bit A 16x32 layout (ISA 7.12.2): lane l holds row m=l%16;
  // element pair p covers k = (p&3)*2 + (p>>2)*16 + (l>>4)*8  (+0,+1)
  const int   m    = lane & 15;
  const int   koff = (lane >> 4) * 8;
  const float* vrow = value + (size_t)(m0 + m) * DVD;
  v16bf areg[16];
#pragma unroll
  for (int kc = 0; kc < 16; ++kc) {
#pragma unroll
    for (int p = 0; p < 8; ++p) {
      const int k = kc * 32 + (p & 3) * 2 + (p >> 2) * 16 + koff;
      float2 f = *(const float2*)(vrow + k);
      areg[kc][2 * p]     = (__bf16)f.x;
      areg[kc][2 * p + 1] = (__bf16)f.y;
    }
  }

  float svec[8] = {0.f, 0.f, 0.f, 0.f, 0.f, 0.f, 0.f, 0.f};

#ifdef USE_TDM
  const unsigned int ldsA = (unsigned int)(uintptr_t)(void*)&Bs[0][0][0];
  const unsigned int ldsB = (unsigned int)(uintptr_t)(void*)&Bs[1][0][0];
  if (wave == 0) tdm_load_wk_tile(wkbf, 0, ldsA);      // prologue: tile 0
#endif

  for (int t = 0; t < 32; ++t) {
    const int n0 = t * 16;
#ifdef USE_TDM
    if (wave == 0) {
      if (t + 1 < 32) {
        tdm_load_wk_tile(wkbf, (t + 1) * 16, ((t + 1) & 1) ? ldsB : ldsA);
        // in-order TENSORcnt: <=1 outstanding means tile t has landed
        __builtin_amdgcn_s_wait_tensorcnt(1);
      } else {
        __builtin_amdgcn_s_wait_tensorcnt(0);
      }
    }
    __syncthreads();
#else
    __syncthreads();
    for (int i = tid; i < 512 * 16; i += 256) {
      const int k = i >> 4, c = i & 15;
      Bs[0][k][c] = wkbf[(size_t)k * DHD + n0 + c];
    }
    __syncthreads();
#endif

    const __bf16 (*Bcur)[24] = Bs[t & 1];
#ifndef USE_TDM
    Bcur = Bs[0];
#endif

    v8f acc = {0.f, 0.f, 0.f, 0.f, 0.f, 0.f, 0.f, 0.f};
#pragma unroll
    for (int kc = 0; kc < 16; ++kc) {
      // 16-bit B 32x16 layout: lane l holds column n=l%16; element e covers k = (e&7) + (e>>3)*16 + (l>>4)*8
      v16bf bf;
      const __bf16* brow = &Bcur[kc * 32 + (lane >> 4) * 8][lane & 15];
#pragma unroll
      for (int e = 0; e < 16; ++e) bf[e] = brow[((e & 7) + (e & 8) * 2) * 24];
      acc = __builtin_amdgcn_wmma_f32_16x16x32_bf16(
          false, areg[kc], false, bf, (short)0, acc, false, false);
    }

    // C layout: acc[r] is row (r + 8*(lane>>4)), col (lane&15) of this N-tile
    const int   h  = n0 + (lane & 15);
    const float qh = qs[h];
    const float wo = wos[h];
#pragma unroll
    for (int r = 0; r < 8; ++r)
      svec[r] += tanhf(qh + acc[r]) * wo;   // TANH is a CDNA5 trans op

    __syncthreads();  // all waves done with Bs[t&1] before it is re-filled
  }

  // reduce the 16 lanes holding each row; lanes 0 / 16 write rows 0-7 / 8-15
  const float bo = bo_p[0];
#pragma unroll
  for (int r = 0; r < 8; ++r) {
    float s = svec[r];
    s += __shfl_xor(s, 1);
    s += __shfl_xor(s, 2);
    s += __shfl_xor(s, 4);
    s += __shfl_xor(s, 8);
    if ((lane & 15) == 0) {
      const int gr = m0 + r + (lane >> 4) * 8;
      const float sc = s + bo;
      scores[gr] = mask[gr] ? sc : -1e9f;
    }
  }
}

// ---------------------------------------------------------------------------
// Kernel 2: in-place masked softmax over S per batch (fixed-order reductions)
// ---------------------------------------------------------------------------
__global__ void softmax_kernel(float* __restrict__ att) {
  __shared__ float red[256];
  const int b = blockIdx.x, tid = threadIdx.x;
  float* s = att + (size_t)b * SEQ;

  float mx = -3.402823466e38f;
  for (int i = tid; i < SEQ; i += 256) mx = fmaxf(mx, s[i]);
  red[tid] = mx; __syncthreads();
  for (int st = 128; st > 0; st >>= 1) {
    if (tid < st) red[tid] = fmaxf(red[tid], red[tid + st]);
    __syncthreads();
  }
  mx = red[0];
  __syncthreads();

  float loc = 0.f;
  for (int i = tid; i < SEQ; i += 256) {
    const float e = expf(s[i] - mx);
    s[i] = e;
    loc += e;
  }
  red[tid] = loc; __syncthreads();
  for (int st = 128; st > 0; st >>= 1) {
    if (tid < st) red[tid] += red[tid + st];
    __syncthreads();
  }
  const float inv = 1.f / red[0];
  for (int i = tid; i < SEQ; i += 256) s[i] *= inv;
}

// ---------------------------------------------------------------------------
// Kernel 3a: partial ctx over S-chunks (deterministic, no float atomics)
// grid (32, 16, 2): batch, S-chunk of 256, column half of 256
// ---------------------------------------------------------------------------
__global__ void ctx_partial_kernel(const float* __restrict__ value,
                                   const float* __restrict__ att,
                                   float* __restrict__ partial) {
  const int b  = blockIdx.x;
  const int ch = blockIdx.y;
  const int v  = blockIdx.z * 256 + threadIdx.x;
  const float* vp = value + ((size_t)b * SEQ + (size_t)ch * 256) * DVD + v;
  const float* ap = att + (size_t)b * SEQ + (size_t)ch * 256;
  float acc = 0.f;
#pragma unroll 4
  for (int s = 0; s < 256; ++s) {
    __builtin_prefetch(vp + (size_t)(s + 16) * DVD, 0, 0);  // global_prefetch_b8
    acc = fmaf(ap[s], vp[(size_t)s * DVD], acc);
  }
  partial[((size_t)b * 16 + ch) * DVD + v] = acc;
}

// Kernel 3b: sum the 16 chunk partials in fixed order -> ctx
__global__ void ctx_reduce_kernel(const float* __restrict__ partial,
                                  float* __restrict__ ctx) {
  const int idx = blockIdx.x * 256 + threadIdx.x;  // 0..16383
  const int b = idx >> 9, v = idx & 511;
  float acc = 0.f;
#pragma unroll
  for (int c = 0; c < 16; ++c)
    acc += partial[((size_t)b * 16 + c) * DVD + v];
  ctx[idx] = acc;
}

// ---------------------------------------------------------------------------
extern "C" void kernel_launch(void* const* d_in, const int* in_sizes, int n_in,
                              void* d_out, int out_size, void* d_ws, size_t ws_size,
                              hipStream_t stream) {
  const float*         query = (const float*)d_in[0];
  const float*         value = (const float*)d_in[1];
  const unsigned char* mask  = (const unsigned char*)d_in[2];
  const float*         Wk    = (const float*)d_in[3];
  const float*         Wq    = (const float*)d_in[4];
  const float*         bq    = (const float*)d_in[5];
  const float*         Wo    = (const float*)d_in[6];
  const float*         bo    = (const float*)d_in[7];

  float* out = (float*)d_out;
  float* ctx = out;                         // [32][512]   (first return value)
  float* att = out + NBATCH * DVD;          // [32][4096]  (second return value)

  float*  qb      = (float*)d_ws;                       // [32][512]
  float*  partial = qb + NBATCH * DHD;                  // [32][16][512]
  __bf16* wkbf    = (__bf16*)(partial + NBATCH * 16 * DVD);  // [512][512] bf16

  qproj_kernel<<<NBATCH, 256, 0, stream>>>(query, Wq, bq, qb);
  wk_bf16_kernel<<<256, 256, 0, stream>>>(Wk, wkbf);
  score_kernel<<<(NBATCH * SEQ) / 128, 256, 0, stream>>>(value, mask, wkbf, Wo, bo, qb, att);
  softmax_kernel<<<NBATCH, 256, 0, stream>>>(att);
  ctx_partial_kernel<<<dim3(NBATCH, 16, 2), 256, 0, stream>>>(value, att, partial);
  ctx_reduce_kernel<<<(NBATCH * DVD) / 256, 256, 0, stream>>>(partial, ctx);
}
